// MultiHeadAttention_59700045414411
// MI455X (gfx1250) — matrix operand order repaired and checked by reading, not compiled, not run on hardware
//
#include <hip/hip_runtime.h>
#include <hip/hip_bf16.h>

// ---------------------------------------------------------------------------
// MHA forward for MI455X (gfx1250, wave32, WMMA 16x16x32 bf16, TDM staging)
//   proj:   Q'=qWq+bq, K', V'   (bf16, head-major (H,B,T,64))
//   attn:   flash attention per (h,b); K/V tiles double-buffered via the
//           Tensor Data Mover (tensor_load_to_lds + s_wait_tensorcnt)
//   ln:     residual + torch-variant LayerNorm (unbiased std, eps on std)
// ---------------------------------------------------------------------------

typedef __attribute__((ext_vector_type(16))) __bf16 v16bf;
typedef __attribute__((ext_vector_type(8)))  __bf16 v8bf;
typedef __attribute__((ext_vector_type(8)))  float  v8f;
typedef __attribute__((ext_vector_type(4)))  unsigned int v4u;
typedef __attribute__((ext_vector_type(8)))  int v8i;
typedef __attribute__((ext_vector_type(4)))  int v4i;

constexpr int kB  = 4;
constexpr int kT  = 2048;
constexpr int kD  = 1024;
constexpr int kH  = 16;
constexpr int kDH = 64;
constexpr int kRows = kB * kT;                       // 8192
constexpr size_t kHM = (size_t)kH * kB * kT * kDH;   // elements per projected tensor

// A-fragment (16x32 bf16) loader from row-major LDS tile with 32-element rows.
// Layout used here (contiguous k, B frags match): lanes {0-15,16-31} hold M=lane%16; element e -> K = e+16*half
__device__ inline v16bf load_a_frag(const __bf16* rowbase, int half) {
  union { v16bf v; v8bf h[2]; } u;
  u.h[0] = *(const v8bf*)(rowbase + 16 * half);
  u.h[1] = *(const v8bf*)(rowbase + 8 + 16 * half);
  return u.v;
}

__device__ inline v8f wmma_bf16(v16bf a, v16bf b, v8f c) {
  // (neg_a, A, neg_b, B, c_mod, C, reuse_a, reuse_b)
  return __builtin_amdgcn_wmma_f32_16x16x32_bf16(false, a, false, b, (short)0, c,
                                                 false, false);
}

// TDM: 2-D tile load (bf16 elements) global -> LDS.  D# per ISA 8.3/8.4:
//   group0: count=1, lds_addr, global_addr[56:0], type=2
//   group1: data_size=2B, tensor_dim0/1, tile_dim0/1, dim0 stride
// amdgpu-toolchain (clang-23) builtin arity: (v4u, v8i, v4i, v4i, v8i, i32 cpol)
__device__ inline void tdm_load_tile_2d(unsigned lds_addr, const void* gptr,
                                        unsigned tile_d0, unsigned tile_d1,
                                        unsigned tensor_d0, unsigned tensor_d1,
                                        unsigned stride0_elems) {
  const unsigned long long ga = (unsigned long long)(uintptr_t)gptr;
  v4u g0;
  g0.x = 1u;                                               // count=1, user mode
  g0.y = lds_addr;                                         // LDS byte address
  g0.z = (unsigned)ga;                                     // global_addr lo
  g0.w = (unsigned)((ga >> 32) & 0x01FFFFFFu) | 0x80000000u;  // hi | type=2
  v8i g1;
  g1[0] = (int)(1u << 16);                                 // data_size = 2 bytes
  g1[1] = (int)(tensor_d0 << 16);                          // tensor_dim0[15:0]
  g1[2] = (int)((tensor_d0 >> 16) | (tensor_d1 << 16));    // dim0 hi | dim1 lo
  g1[3] = (int)((tensor_d1 >> 16) | (tile_d0 << 16));      // dim1 hi | tile_dim0
  g1[4] = (int)(tile_d1 & 0xFFFFu);                        // tile_dim1, tile_dim2=0
  g1[5] = (int)stride0_elems;                              // tensor_dim0_stride lo
  g1[6] = 0;                                               // stride0 hi | stride1 lo
  g1[7] = 0;                                               // stride1 hi
  v4i g2 = {0, 0, 0, 0};
  v4i g3 = {0, 0, 0, 0};
  v8i g4 = {0, 0, 0, 0, 0, 0, 0, 0};
  __builtin_amdgcn_tensor_load_to_lds(g0, g1, g2, g3, g4, 0);
}

// ---------------------------------------------------------------------------
// Kernel 1: projection GEMM  (M=8192, N=1024, K=1024), z selects q/k/v matrix.
// WG = 128 threads (4 waves); 128x64 tile; K-step 32; f32->bf16 on LDS staging.
// Each wave computes 32x64 (2 A-frags x 4 B-frags = 8 WMMAs / K-step).
// ---------------------------------------------------------------------------
__global__ __launch_bounds__(128)
void proj_gemm_kernel(const float* __restrict__ qin, const float* __restrict__ kin,
                      const float* __restrict__ vin,
                      const float* __restrict__ Wq, const float* __restrict__ Wk,
                      const float* __restrict__ Wv,
                      const float* __restrict__ bq, const float* __restrict__ bk,
                      const float* __restrict__ bv,
                      __bf16* __restrict__ Qh, __bf16* __restrict__ Kh,
                      __bf16* __restrict__ Vh) {
  const float* act;  const float* W;  const float* bias;  __bf16* out;
  if (blockIdx.z == 0)      { act = qin; W = Wq; bias = bq; out = Qh; }
  else if (blockIdx.z == 1) { act = kin; W = Wk; bias = bk; out = Kh; }
  else                      { act = vin; W = Wv; bias = bv; out = Vh; }

  __shared__ __align__(16) __bf16 sA[128 * 32];  // [m][k]   8KB
  __shared__ __align__(16) __bf16 sBt[64 * 32];  // [n][k]   4KB (K-transposed)

  const int tid  = threadIdx.x;
  const int lane = tid & 31;
  const int wave = tid >> 5;
  const int half = lane >> 4;
  const int ln16 = lane & 15;

  const int mbase = blockIdx.x * 128;
  const int nbase = blockIdx.y * 64;

  v8f acc[2][4];
#pragma unroll
  for (int mi = 0; mi < 2; ++mi)
#pragma unroll
    for (int j = 0; j < 4; ++j)
#pragma unroll
      for (int i = 0; i < 8; ++i) acc[mi][j][i] = 0.0f;

  for (int kk = 0; kk < kD; kk += 32) {
    __syncthreads();
    {  // stage A: 128x32, thread tid converts one 32-elem row f32 -> bf16
      const float* src = act + (size_t)(mbase + tid) * kD + kk;
      __bf16* dst = &sA[tid * 32];
#pragma unroll
      for (int i = 0; i < 32; ++i) dst[i] = (__bf16)src[i];
      if (kk + 32 < kD) __builtin_prefetch(src + 32, 0, 0);
    }
    {  // stage B transposed: W[kk+kr][nbase+n] -> sBt[n*32 + kr]
      const int kr = tid >> 2;
      const int ns = (tid & 3) * 16;
      const float* src = W + (size_t)(kk + kr) * kD + nbase + ns;
#pragma unroll
      for (int i = 0; i < 16; ++i) sBt[(ns + i) * 32 + kr] = (__bf16)src[i];
      if (kk + 32 < kD) __builtin_prefetch(src + 32 * kD, 0, 0);
    }
    __syncthreads();

    v16bf bfr[4];
#pragma unroll
    for (int j = 0; j < 4; ++j)
      bfr[j] = *(const v16bf*)&sBt[(j * 16 + ln16) * 32 + 16 * half];
#pragma unroll
    for (int mi = 0; mi < 2; ++mi) {
      const v16bf a = load_a_frag(&sA[(wave * 32 + mi * 16 + ln16) * 32], half);
#pragma unroll
      for (int j = 0; j < 4; ++j) acc[mi][j] = wmma_bf16(a, bfr[j], acc[mi][j]);
    }
  }

  // epilogue: + bias, write bf16 head-major (H,B,T,64)
#pragma unroll
  for (int mi = 0; mi < 2; ++mi) {
    const int row0 = mbase + wave * 32 + mi * 16;
#pragma unroll
    for (int j = 0; j < 4; ++j) {
      const int d     = nbase + j * 16 + ln16;
      const float bv_ = bias[d];
      const int hh = d >> 6, dh = d & 63;
#pragma unroll
      for (int i = 0; i < 8; ++i) {
        const int r  = row0 + i + 8 * half;
        const int bb = r / kT, tt = r % kT;
        const size_t idx = (((size_t)(hh * kB + bb) * kT) + tt) * kDH + dh;
        out[idx] = (__bf16)(acc[mi][j][i] + bv_);
      }
    }
  }
}

// ---------------------------------------------------------------------------
// Kernel 2: flash attention per (query-tile, b, h). WG = 128 (4 waves), each
// wave owns a 16x64 output tile; loop over 2048 keys in blocks of 32.
// K/V tiles are double-buffered through the Tensor Data Mover: wave 0 issues
// the next block's DMAs right after the barrier retiring the previous block's
// readers, and only waits TENSORcnt for the buffer about to be consumed.
// ---------------------------------------------------------------------------
__global__ __launch_bounds__(128)
void attn_kernel(const __bf16* __restrict__ Qh, const __bf16* __restrict__ Kh,
                 const __bf16* __restrict__ Vh, float* __restrict__ attn_out) {
  __shared__ __align__(16) __bf16 sK[2][32 * 64];   // [buf][key][dh]  (TDM dest)
  __shared__ __align__(16) __bf16 sV[2][32 * 64];   // [buf][key][dh]  (TDM dest)
  __shared__ __align__(16) __bf16 sVt[64 * 32];     // [dh][key]       (transposed)
  __shared__ __align__(16) __bf16 sP[4 * 16 * 32];  // per-wave [row][key]

  const int tid  = threadIdx.x;
  const int lane = tid & 31;
  const int wave = tid >> 5;
  const int half = lane >> 4;
  const int ln16 = lane & 15;

  const int qt = blockIdx.x;       // 64-query tile
  const int bb = blockIdx.y;       // batch
  const int hh = blockIdx.z;       // head

  const size_t hbrow = (size_t)(hh * kB + bb) * kT;
  const __bf16* Qbase = Qh + hbrow * kDH;
  const __bf16* Kbase = Kh + hbrow * kDH;
  const __bf16* Vbase = Vh + hbrow * kDH;

  const unsigned sK_lds[2] = {(unsigned)(uintptr_t)&sK[0][0],
                              (unsigned)(uintptr_t)&sK[1][0]};
  const unsigned sV_lds[2] = {(unsigned)(uintptr_t)&sV[0][0],
                              (unsigned)(uintptr_t)&sV[1][0]};

  const int qrow = qt * 64 + wave * 16 + ln16;  // this lane's query row (A layout)

  // preload Q A-fragments: f=0 -> dh 0..31, f=1 -> dh 32..63
  v16bf aQ[2];
#pragma unroll
  for (int f = 0; f < 2; ++f)
#pragma unroll
    for (int e = 0; e < 16; ++e) {
      const int dh = e + 16 * half + 32 * f;
      aQ[f][e] = Qbase[(size_t)qrow * kDH + dh];
    }

  v8f O[4];
  float m[8], l[8];
#pragma unroll
  for (int i = 0; i < 8; ++i) { m[i] = -1e30f; l[i] = 0.0f; }
#pragma unroll
  for (int j = 0; j < 4; ++j)
#pragma unroll
    for (int i = 0; i < 8; ++i) O[j][i] = 0.0f;

  const float scale = 0.125f;  // 1/sqrt(64)
  constexpr int kNB = kT / 32; // 64 key blocks

  // prologue: DMA first key block into buffer 0
  if (wave == 0) {
    tdm_load_tile_2d(sK_lds[0], Kbase, kDH, 32, kDH, kT, kDH);
    tdm_load_tile_2d(sV_lds[0], Vbase, kDH, 32, kDH, kT, kDH);
  }

  for (int kb = 0; kb < kNB; ++kb) {
    const int cur = kb & 1;
    if (wave == 0) {
      // only buf[cur]'s two loads are outstanding here -> precise wait
      __builtin_amdgcn_s_wait_tensorcnt(0);
    }
    __syncthreads();  // publish buf[cur]; all waves done reading buf[1-cur]

    if (wave == 0 && kb + 1 < kNB) {  // prefetch next block under this block's compute
      tdm_load_tile_2d(sK_lds[1 - cur], Kbase + (size_t)(kb + 1) * 32 * kDH,
                       kDH, 32, kDH, kT, kDH);
      tdm_load_tile_2d(sV_lds[1 - cur], Vbase + (size_t)(kb + 1) * 32 * kDH,
                       kDH, 32, kDH, kT, kDH);
    }

    {  // transpose V: sV[cur][key][dh] -> sVt[dh][key] (all 128 threads)
      const int key = tid >> 2;
      const int dhs = (tid & 3) * 16;
      const v8bf a0 = *(const v8bf*)&sV[cur][key * 64 + dhs];
      const v8bf a1 = *(const v8bf*)&sV[cur][key * 64 + dhs + 8];
#pragma unroll
      for (int i = 0; i < 8; ++i) sVt[(dhs + i) * 32 + key] = a0[i];
#pragma unroll
      for (int i = 0; i < 8; ++i) sVt[(dhs + 8 + i) * 32 + key] = a1[i];
    }
    __syncthreads();

    // scores S = Q Kt : two 16-key tiles, each 2 WMMAs over dh
    v8f s0, s1;
#pragma unroll
    for (int i = 0; i < 8; ++i) { s0[i] = 0.0f; s1[i] = 0.0f; }
    {
      const v16bf b00 = *(const v16bf*)&sK[cur][(0 * 16 + ln16) * 64 + 16 * half];
      const v16bf b01 = *(const v16bf*)&sK[cur][(0 * 16 + ln16) * 64 + 32 + 16 * half];
      s0 = wmma_bf16(aQ[0], b00, s0);
      s0 = wmma_bf16(aQ[1], b01, s0);
      const v16bf b10 = *(const v16bf*)&sK[cur][(1 * 16 + ln16) * 64 + 16 * half];
      const v16bf b11 = *(const v16bf*)&sK[cur][(1 * 16 + ln16) * 64 + 32 + 16 * half];
      s1 = wmma_bf16(aQ[0], b10, s1);
      s1 = wmma_bf16(aQ[1], b11, s1);
    }

    // online softmax per row (rows i+8*half live across the 16 lanes of a half)
#pragma unroll
    for (int i = 0; i < 8; ++i) {
      const float v0 = s0[i] * scale;
      const float v1 = s1[i] * scale;
      float mloc = fmaxf(v0, v1);
      mloc = fmaxf(mloc, __shfl_xor(mloc, 1));
      mloc = fmaxf(mloc, __shfl_xor(mloc, 2));
      mloc = fmaxf(mloc, __shfl_xor(mloc, 4));
      mloc = fmaxf(mloc, __shfl_xor(mloc, 8));
      const float mnew  = fmaxf(m[i], mloc);
      const float alpha = __expf(m[i] - mnew);
      const float p0 = __expf(v0 - mnew);
      const float p1 = __expf(v1 - mnew);
      float ps = p0 + p1;
      ps += __shfl_xor(ps, 1);
      ps += __shfl_xor(ps, 2);
      ps += __shfl_xor(ps, 4);
      ps += __shfl_xor(ps, 8);
      l[i] = l[i] * alpha + ps;
      m[i] = mnew;
#pragma unroll
      for (int j = 0; j < 4; ++j) O[j][i] *= alpha;
      // park P (bf16) in per-wave LDS: C-layout -> row-major [16][32]
      const int r = i + 8 * half;
      sP[wave * 512 + r * 32 + ln16]      = (__bf16)p0;
      sP[wave * 512 + r * 32 + 16 + ln16] = (__bf16)p1;
    }

    // O += P x V  (A = P 16x32, B = V 32x16 per dh tile; sVt gives contiguous B)
    const v16bf pa = load_a_frag(&sP[wave * 512 + ln16 * 32], half);
#pragma unroll
    for (int j = 0; j < 4; ++j) {
      const v16bf vb = *(const v16bf*)&sVt[(j * 16 + ln16) * 32 + 16 * half];
      O[j] = wmma_bf16(pa, vb, O[j]);
    }
  }

  // finalize: divide by l, write f32 merged-head layout (B,T,D)
#pragma unroll
  for (int j = 0; j < 4; ++j) {
    const int dh = j * 16 + ln16;
#pragma unroll
    for (int i = 0; i < 8; ++i) {
      const int t = qt * 64 + wave * 16 + i + 8 * half;
      attn_out[((size_t)bb * kT + t) * kD + hh * kDH + dh] = O[j][i] / l[i];
    }
  }
}

// ---------------------------------------------------------------------------
// Kernel 3: out = LayerNorm(attn + q)   (unbiased var over D-1, eps on std)
// ---------------------------------------------------------------------------
__global__ __launch_bounds__(256)
void ln_kernel(const float* __restrict__ attn, const float* __restrict__ qin,
               const float* __restrict__ gamma, const float* __restrict__ beta,
               float* __restrict__ out) {
  __shared__ float red[256];
  const int row = blockIdx.x;
  const int tid = threadIdx.x;

  float x[4];
  float s = 0.0f;
#pragma unroll
  for (int i = 0; i < 4; ++i) {
    const int c = i * 256 + tid;
    x[i] = attn[(size_t)row * kD + c] + qin[(size_t)row * kD + c];
    s += x[i];
  }
  red[tid] = s;
  __syncthreads();
  for (int off = 128; off > 0; off >>= 1) {
    if (tid < off) red[tid] += red[tid + off];
    __syncthreads();
  }
  const float mean = red[0] / (float)kD;
  __syncthreads();

  float vs = 0.0f;
#pragma unroll
  for (int i = 0; i < 4; ++i) {
    const float d = x[i] - mean;
    vs += d * d;
  }
  red[tid] = vs;
  __syncthreads();
  for (int off = 128; off > 0; off >>= 1) {
    if (tid < off) red[tid] += red[tid + off];
    __syncthreads();
  }
  const float var = red[0] / (float)(kD - 1);
  const float inv = 1.0f / (sqrtf(var) + 1e-8f);

#pragma unroll
  for (int i = 0; i < 4; ++i) {
    const int c = i * 256 + tid;
    out[(size_t)row * kD + c] = gamma[c] * (x[i] - mean) * inv + beta[c];
  }
}

// ---------------------------------------------------------------------------
extern "C" void kernel_launch(void* const* d_in, const int* in_sizes, int n_in,
                              void* d_out, int out_size, void* d_ws, size_t ws_size,
                              hipStream_t stream) {
  const float* q     = (const float*)d_in[0];
  const float* k     = (const float*)d_in[1];
  const float* v     = (const float*)d_in[2];
  const float* Wq    = (const float*)d_in[3];
  const float* bq    = (const float*)d_in[4];
  const float* Wk    = (const float*)d_in[5];
  const float* bk    = (const float*)d_in[6];
  const float* Wv    = (const float*)d_in[7];
  const float* bv    = (const float*)d_in[8];
  const float* gamma = (const float*)d_in[9];
  const float* beta  = (const float*)d_in[10];

  __bf16* Qh = (__bf16*)d_ws;
  __bf16* Kh = Qh + kHM;
  __bf16* Vh = Kh + kHM;
  float*  attn = (float*)(Vh + kHM);

  proj_gemm_kernel<<<dim3(kRows / 128, kD / 64, 3), 128, 0, stream>>>(
      q, k, v, Wq, Wk, Wv, bq, bk, bv, Qh, Kh, Vh);
  attn_kernel<<<dim3(kT / 64, kB, kH), 128, 0, stream>>>(Qh, Kh, Vh, attn);
  ln_kernel<<<dim3(kRows), 256, 0, stream>>>(attn, q, gamma, beta, (float*)d_out);
}
